// FusedLayer_49048526520369
// MI455X (gfx1250) — compile-verified, matrix operand order checked by reading, not run on hardware
//
#include <hip/hip_runtime.h>
#include <hip/hip_bf16.h>

// ---------------------------------------------------------------------------
// 4 fused Conv2d+Bias+ReLU branches as implicit GEMM on CDNA5 WMMA bf16.
// fp32 in/out, bf16 multiply / f32 accumulate (v_wmma_f32_16x16x32_bf16).
//
//   out[M=N*H*W, N=Cout] = im2col(x)[M, K] * W^T[K, N]
//   K-order (kh, kw, ci): kb-outer loop over kernel taps, fully unrolled
//   ci-inner loop -> all global loads use compile-time immediate offsets
//   from a per-tap base pointer (channel offsets < 8MB 24-bit IOFFSET).
//
//   M = 100352 = 784*128 exact, Cout multiples of 64, Cin = 256 fixed.
//
// Pipeline: ping-pong LDS tiles (one s_barrier per K-step) + global->VGPR
// register prefetch of the next K-tile issued under the current WMMA work.
// f32->bf16 packing via hardware v_cvt_pk_bf16_f32 (inline asm; no clang
// builtin on this toolchain).
// ---------------------------------------------------------------------------

typedef __attribute__((ext_vector_type(16))) __bf16 v16bf;
typedef __attribute__((ext_vector_type(8)))  float  v8f;

#define CONV_H   56
#define CONV_W   56
#define CONV_HW  3136
#define CONV_B   32
#define CTOT     512
#define CIN      256

#define BM 128           // block tile rows (spatial)
#define BN 64            // block tile cols (out channels)
#define BK 32            // K step (matches wmma 16x16x32 bf16)
#define NCI (CIN / BK)   // 8 ci-blocks per kernel tap

#if defined(__has_builtin)
#if __has_builtin(__builtin_amdgcn_cvt_pk_bf16_f32)
#define HAVE_PK_BF16_BUILTIN 1
#endif
#endif

__device__ __forceinline__ unsigned int pack_bf(float lo, float hi) {
#ifdef HAVE_PK_BF16_BUILTIN
    typedef __attribute__((ext_vector_type(2))) __bf16 v2bf;
    v2bf p = __builtin_amdgcn_cvt_pk_bf16_f32(lo, hi);
    return __builtin_bit_cast(unsigned int, p);
#else
    // CDNA5 hardware packed convert (RNE): dst[15:0]=bf16(lo), [31:16]=bf16(hi)
    unsigned int r;
    asm("v_cvt_pk_bf16_f32 %0, %1, %2" : "=v"(r) : "v"(lo), "v"(hi));
    return r;
#endif
}

union AFrag { unsigned int u[8]; v16bf v; };
union CAcc  { float f[8];        v8f  v; };

template <int KS>
__global__ __launch_bounds__(256)
void conv_bias_relu_igemm(const float* __restrict__ x,
                          const float* __restrict__ wgt,
                          const float* __restrict__ bias,
                          float* __restrict__ out,
                          int c_off)
{
    constexpr int KSQ = KS * KS;
    constexpr int PAD = (KS == 3) ? 1 : 0;
    constexpr int KTOT = CIN * KSQ;

    // ping-pong K-tiles; +1 pad on trailing dim -> conflict-free ds access
    __shared__ unsigned int sA[2][BM][BK / 2 + 1];   // 2 x 128 x 17 pairs
    __shared__ unsigned int sB[2][BN][BK / 2 + 1];   // 2 x  64 x 17 pairs
    __shared__ float        sC[BM][BN + 2];          // 128 x 66 f32 epilogue

    const int tid  = threadIdx.x;
    const int lane = tid & 31;
    const int wave = tid >> 5;        // 8 waves
    const int wm   = wave >> 1;       // 0..3 : 32 output rows each
    const int wn   = wave & 1;        // 0..1 : 32 output cols each
    const int half = lane >> 4;       // ISA 16-bit operand half selector
    const int ml   = lane & 15;

    const int gm0 = blockIdx.x * BM;
    const int c0  = blockIdx.y * BN;

    // ---- loop-invariant per-thread im2col coordinates -------------------
    // A pairs: q = tid + 256t -> m = tid&127 (t-invariant), kp = (tid>>7)+2t
    const int mA  = tid & 127;
    const int kpA = tid >> 7;             // 0 or 1
    const int gm  = gm0 + mA;
    const int bN  = gm / CONV_HW;
    const int hwm = gm - bN * CONV_HW;
    const int hA  = hwm / CONV_W;
    const int wA  = hwm - hA * CONV_W;
    // fold batch plane and kpA channel offset into the A base pointer
    const float* __restrict__ xb =
        x + (size_t)bN * (CIN * CONV_HW) + 2 * kpA * CONV_HW;

    // B pairs: kpB = tid&15 (t-invariant), n = (tid>>4) + 16t
    const int kpB = tid & 15;
    const int nB  = tid >> 4;
    const float* __restrict__ wBase[4];
#pragma unroll
    for (int t = 0; t < 4; ++t)
        wBase[t] = wgt + (size_t)(c0 + nB + 16 * t) * KTOT + (2 * kpB) * KSQ;

    CAcc acc[2][2];
#pragma unroll
    for (int i = 0; i < 2; ++i)
#pragma unroll
        for (int j = 0; j < 2; ++j)
#pragma unroll
            for (int r = 0; r < 8; ++r) acc[i][j].f[r] = 0.0f;

    // ---- per-tap state + register-stage prefetch ------------------------
    float aReg[8][2];
    float bReg[4][2];
    const float* pA = xb;           // per-tap A base (incl. halo shift)
    const float* pB[4];
    bool okA = true;
#pragma unroll
    for (int t = 0; t < 4; ++t) pB[t] = wBase[t];

    auto set_kb = [&](int kb) {     // advance per-tap bases / halo predicate
        int kh = 0, kw = 0;
        if (KS == 3) { kh = kb / KS; kw = kb - kh * KS; }
        const int ih = hA + kh - PAD;
        const int iw = wA + kw - PAD;
        okA = (KS == 1) || ((unsigned)ih < (unsigned)CONV_H &&
                            (unsigned)iw < (unsigned)CONV_W);
        pA = xb + ih * CONV_W + iw;
#pragma unroll
        for (int t = 0; t < 4; ++t) pB[t] = wBase[t] + kb;
    };

    auto prefetch_ci = [&](int ci0) {   // ci0 is compile-time in unrolled loop
#pragma unroll
        for (int t = 0; t < 8; ++t) {   // imm offsets: (ci0+4t+e)*HW < 8MB
            aReg[t][0] = okA ? pA[(ci0 + 4 * t) * CONV_HW]     : 0.0f;
            aReg[t][1] = okA ? pA[(ci0 + 4 * t + 1) * CONV_HW] : 0.0f;
        }
#pragma unroll
        for (int t = 0; t < 4; ++t) {
            bReg[t][0] = pB[t][(ci0    ) * KSQ];
            bReg[t][1] = pB[t][(ci0 + 1) * KSQ];
        }
    };

    set_kb(0);
    prefetch_ci(0);

    for (int kb = 0; kb < KSQ; ++kb) {
#pragma unroll
        for (int j = 0; j < NCI; ++j) {          // fully unrolled: ci0, buf
            const int buf = j & 1;               // compile-time ping-pong
            // ---- pack f32->bf16 (v_cvt_pk_bf16_f32), fill LDS tile ------
#pragma unroll
            for (int t = 0; t < 8; ++t)
                sA[buf][mA][kpA + 2 * t] = pack_bf(aReg[t][0], aReg[t][1]);
#pragma unroll
            for (int t = 0; t < 4; ++t)
                sB[buf][nB + 16 * t][kpB] = pack_bf(bReg[t][0], bReg[t][1]);
            __syncthreads();                     // single barrier per K-step

            // ---- issue next tile's global loads (hidden under wmma) -----
            if (j + 1 < NCI) {
                prefetch_ci((j + 1) * BK);
            } else if (kb + 1 < KSQ) {
                set_kb(kb + 1);
                prefetch_ci(0);
            }

            // ---- fragment loads (ISA 7.12.2 16-bit layout) + 4 WMMA -----
            AFrag a[2], bf[2];
#pragma unroll
            for (int s = 0; s < 2; ++s) {
#pragma unroll
                for (int i = 0; i < 8; ++i) {
                    const int kp = (i & 3) + ((i >> 2) << 3) + 4 * half;
                    a[s].u[i]  = sA[buf][wm * 32 + s * 16 + ml][kp];
                    bf[s].u[i] = sB[buf][wn * 32 + s * 16 + ml][kp];
                }
            }
#pragma unroll
            for (int sm = 0; sm < 2; ++sm)
#pragma unroll
                for (int sn = 0; sn < 2; ++sn)
                    acc[sm][sn].v = __builtin_amdgcn_wmma_f32_16x16x32_bf16(
                        false, a[sm].v, false, bf[sn].v,
                        (short)0, acc[sm][sn].v, false, false);
        }
    }

    // ---- epilogue: stage through LDS for coalesced bias+ReLU stores -----
    __syncthreads();
#pragma unroll
    for (int sm = 0; sm < 2; ++sm)
#pragma unroll
        for (int sn = 0; sn < 2; ++sn)
#pragma unroll
            for (int r = 0; r < 8; ++r)
                sC[wm * 32 + sm * 16 + r + 8 * half]
                  [wn * 32 + sn * 16 + ml] = acc[sm][sn].f[r];
    __syncthreads();

#pragma unroll 8
    for (int t = 0; t < 32; ++t) {
        const int e = tid + t * 256;          // 0..8191
        const int m = e & (BM - 1);           // lane-contiguous -> coalesced
        const int n = e >> 7;                 // 0..63, constant per wave
        const int g = gm0 + m;
        const int b = g / CONV_HW;
        const int hw = g - b * CONV_HW;
        float v = sC[m][n] + bias[c0 + n];
        v = fmaxf(v, 0.0f);
        out[((size_t)b * CTOT + (c_off + c0 + n)) * CONV_HW + hw] = v;
    }
}

extern "C" void kernel_launch(void* const* d_in, const int* in_sizes, int n_in,
                              void* d_out, int out_size, void* d_ws, size_t ws_size,
                              hipStream_t stream) {
    (void)in_sizes; (void)n_in; (void)out_size; (void)d_ws; (void)ws_size;

    static const int Cout[4] = {64, 128, 256, 64};
    static const int Ks[4]   = {1, 3, 3, 1};
    static const int Coff[4] = {0, 64, 192, 448};   // concat channel offsets

    float* out = (float*)d_out;
    for (int i = 0; i < 4; ++i) {
        // setup_inputs dict order: x0,w0,b0, x1,w1,b1, ...
        const float* x = (const float*)d_in[3 * i + 0];
        const float* w = (const float*)d_in[3 * i + 1];
        const float* b = (const float*)d_in[3 * i + 2];
        dim3 grid((CONV_B * CONV_HW) / BM, Cout[i] / BN);  // 784 x {1,2,4,1}
        if (Ks[i] == 1)
            conv_bias_relu_igemm<1><<<grid, 256, 0, stream>>>(x, w, b, out, Coff[i]);
        else
            conv_bias_relu_igemm<3><<<grid, 256, 0, stream>>>(x, w, b, out, Coff[i]);
    }
}
